// CrystalGATVAE_79637283602855
// MI455X (gfx1250) — matrix operand order repaired and checked by reading, not compiled, not run on hardware
//
#include <hip/hip_runtime.h>
#include <math.h>

// ---------------------------------------------------------------------------
// CrystalGATVAE forward for gfx1250 (MI455X).
//  - All dense matmuls: v_wmma_f32_16x16x32_bf16, operands pre-converted to
//    zero-padded bf16 (A: [Mp,Kp] row-major, B: [Np,Kp] = B^T) so the GEMM
//    inner loop is branchless: 8x global_load_b128 + 4x v_wmma per k-step.
//  - Graph segment ops via atomics; algebraic shortcuts: degree-scaled npj
//    scatter, rank-1 recon_edge.
// ---------------------------------------------------------------------------

#define NFD   4
#define EFDIM 3
#define NHEAD 4

typedef __bf16 bf16_t;
typedef bf16_t   v16bf __attribute__((ext_vector_type(16)));
typedef float    v8f   __attribute__((ext_vector_type(8)));
typedef unsigned uv8   __attribute__((ext_vector_type(8)));

static __host__ __device__ inline unsigned f2u(float f){ union {float f; unsigned u;} v; v.f=f; return v.u; }
static __host__ __device__ inline float    u2f(unsigned u){ union {float f; unsigned u;} v; v.u=u; return v.f; }
// order-preserving float -> u32 encoding (for atomicMax on floats)
static __host__ __device__ inline unsigned fenc(float f){ unsigned u=f2u(f); return (u & 0x80000000u) ? ~u : (u | 0x80000000u); }
static __host__ __device__ inline float    fdec(unsigned e){ unsigned u = (e & 0x80000000u) ? (e & 0x7fffffffu) : ~e; return u2f(u); }

static __device__ inline unsigned short f2bfu(float f){
  unsigned u = f2u(f);
  u += 0x7fffu + ((u >> 16) & 1u);          // round-to-nearest-even
  return (unsigned short)(u >> 16);
}

static inline unsigned gsz(long long total, int bs){ return (unsigned)((total + bs - 1) / bs); }
static inline int rup32(int x){ return ((x + 31) / 32) * 32; }

// ---------------------------------------------------------------------------
// Operand conversion: fp32 -> zero-padded bf16
// ---------------------------------------------------------------------------
// A [M,K] (ld=lda) -> dst [Mp,Kp] row-major bf16, zero pad
__global__ void cvt_a_kernel(const float* __restrict__ A, unsigned short* __restrict__ dst,
                             int M, int K, int lda, int Kp, long long total)
{
  long long idx = (long long)blockIdx.x * blockDim.x + threadIdx.x;
  if (idx >= total) return;
  int r = (int)(idx / Kp), c = (int)(idx % Kp);
  int rc = min(r, M - 1), cc = min(c, K - 1);
  float v = A[(size_t)rc * lda + cc];
  v = (r < M && c < K) ? v : 0.f;
  dst[idx] = f2bfu(v);
}

// B [K,N] (ld=ldb) -> dst [Np,Kp] row-major bf16 (i.e. B transposed), zero pad
__global__ void cvt_b_kernel(const float* __restrict__ B, unsigned short* __restrict__ dst,
                             int K, int N, int ldb, int Kp, long long total)
{
  long long idx = (long long)blockIdx.x * blockDim.x + threadIdx.x;
  if (idx >= total) return;
  int n = (int)(idx / Kp), k = (int)(idx % Kp);
  int nc = min(n, N - 1), kc = min(k, K - 1);
  float v = B[(size_t)kc * ldb + nc];
  v = (n < N && k < K) ? v : 0.f;
  dst[idx] = f2bfu(v);
}

// ---------------------------------------------------------------------------
// WMMA GEMM on pre-converted bf16:  C[M,N] = post + act(Abf @ Bbf^T + bias)
// One wave computes a 32x32 C tile (2x2 WMMA fragments, 4 accumulators).
// Inner loop: 8x b128 loads + 4x v_wmma_f32_16x16x32_bf16, fully branchless.
// ---------------------------------------------------------------------------
static __device__ inline v16bf ld_frag(const unsigned* __restrict__ p, int off2)
{
  const uint4 lo = *(const uint4*)(p);
  const uint4 hi = *(const uint4*)(p + off2);
  uv8 t;
  t[0] = lo.x; t[1] = lo.y; t[2] = lo.z; t[3] = lo.w;
  t[4] = hi.x; t[5] = hi.y; t[6] = hi.z; t[7] = hi.w;
  return __builtin_bit_cast(v16bf, t);
}

static __device__ inline void store_tile(float* __restrict__ C, int ldc, const v8f& acc,
                                         int rbase, int col, int M, int N,
                                         const float* __restrict__ bias,
                                         const float* __restrict__ post, int act)
{
  if (col >= N) return;
  const float bi = bias ? bias[col] : 0.f;
  const float po = post ? post[col] : 0.f;
#pragma unroll
  for (int r = 0; r < 8; ++r) {
    const int row = rbase + r;
    if (row < M) {
      float v = acc[r] + bi;
      if (act == 1) v = fmaxf(v, 0.f);
      C[(size_t)row * ldc + col] = v + po;
    }
  }
}

__global__ void __launch_bounds__(32)
wmma_gemm_bf(const unsigned short* __restrict__ Abf, const unsigned short* __restrict__ Bbf,
             const float* __restrict__ bias, const float* __restrict__ post,
             float* __restrict__ C, int M, int N, int Kp, int ldc, int act)
{
  const int lane = threadIdx.x & 31;
  const int m0 = blockIdx.x * 32;
  const int n0 = blockIdx.y * 32;
  const int half = lane >> 4;        // 0: lanes 0-15, 1: lanes 16-31
  const int l16  = lane & 15;

  // A fragment lane layout (16x32 bf16): lane covers K = half*8 + {0..7, 16..23}
  // -> as u32: base + half*4, two uint4 at +0 and +8.
  const unsigned* a0 = (const unsigned*)(Abf + (size_t)(m0 + l16)      * Kp) + half * 4;
  const unsigned* a1 = (const unsigned*)(Abf + (size_t)(m0 + 16 + l16) * Kp) + half * 4;
  // B fragment lane layout (32x16 bf16): lane covers K = half*8 + {0..7, 16..23}
  // -> as u32: base + half*4, two uint4 at +0 and +8.
  const unsigned* b0 = (const unsigned*)(Bbf + (size_t)(n0 + l16)      * Kp) + half * 4;
  const unsigned* b1 = (const unsigned*)(Bbf + (size_t)(n0 + 16 + l16) * Kp) + half * 4;

  v8f acc00 = {0.f,0.f,0.f,0.f,0.f,0.f,0.f,0.f};
  v8f acc01 = acc00, acc10 = acc00, acc11 = acc00;

  const int kuEnd = Kp >> 1;         // u32 per row
  for (int ku = 0; ku < kuEnd; ku += 16) {
    const v16bf af0 = ld_frag(a0 + ku, 8);
    const v16bf af1 = ld_frag(a1 + ku, 8);
    const v16bf bf0 = ld_frag(b0 + ku, 8);
    const v16bf bf1 = ld_frag(b1 + ku, 8);
    acc00 = __builtin_amdgcn_wmma_f32_16x16x32_bf16(false, af0, false, bf0, (short)0, acc00, false, false);
    acc01 = __builtin_amdgcn_wmma_f32_16x16x32_bf16(false, af0, false, bf1, (short)0, acc01, false, false);
    acc10 = __builtin_amdgcn_wmma_f32_16x16x32_bf16(false, af1, false, bf0, (short)0, acc10, false, false);
    acc11 = __builtin_amdgcn_wmma_f32_16x16x32_bf16(false, af1, false, bf1, (short)0, acc11, false, false);
  }

  // C/D layout: vgpr r -> row (half*8 + r), col = l16 (+16 for the second n tile)
  const int rb0 = m0 + half * 8;
  const int rb1 = m0 + 16 + half * 8;
  const int c0  = n0 + l16;
  const int c1  = n0 + 16 + l16;
  store_tile(C, ldc, acc00, rb0, c0, M, N, bias, post, act);
  store_tile(C, ldc, acc01, rb0, c1, M, N, bias, post, act);
  store_tile(C, ldc, acc10, rb1, c0, M, N, bias, post, act);
  store_tile(C, ldc, acc11, rb1, c1, M, N, bias, post, act);
}

// ---------------------------------------------------------------------------
// Utility fills
// ---------------------------------------------------------------------------
__global__ void fill_f32(float* p, float v, long long n){
  long long i = (long long)blockIdx.x * blockDim.x + threadIdx.x;
  if (i < n) p[i] = v;
}
__global__ void fill_u32(unsigned* p, unsigned v, long long n){
  long long i = (long long)blockIdx.x * blockDim.x + threadIdx.x;
  if (i < n) p[i] = v;
}

// ---------------------------------------------------------------------------
// Edge MLP: e = relu(ef @ em_w1 + b1) @ em_w2 + b2    [E,3] -> [E,3]
// ---------------------------------------------------------------------------
__global__ void edge_mlp_kernel(const float* __restrict__ ef,
                                const float* __restrict__ w1, const float* __restrict__ b1,
                                const float* __restrict__ w2, const float* __restrict__ b2,
                                float* __restrict__ out, int E)
{
  int e = blockIdx.x * blockDim.x + threadIdx.x;
  if (e >= E) return;
  float x0 = ef[(size_t)e*3+0], x1 = ef[(size_t)e*3+1], x2 = ef[(size_t)e*3+2];
  float h[6];
#pragma unroll
  for (int j = 0; j < 6; ++j) {
    float a = b1[j] + x0*w1[0*6+j] + x1*w1[1*6+j] + x2*w1[2*6+j];
    h[j] = a > 0.f ? a : 0.f;
  }
#pragma unroll
  for (int j = 0; j < 3; ++j) {
    float a = b2[j];
#pragma unroll
    for (int k = 0; k < 6; ++k) a += h[k]*w2[k*3+j];
    out[(size_t)e*3+j] = a;
  }
}

// degree count (node appears once per incident edge endpoint)
__global__ void degree_kernel(const int* __restrict__ src, const int* __restrict__ dst,
                              unsigned* __restrict__ deg, int E)
{
  int e = blockIdx.x * blockDim.x + threadIdx.x;
  if (e >= E) return;
  atomicAdd(&deg[src[e]], 1u);
  atomicAdd(&deg[dst[e]], 1u);
}

// tmp[v, c] = (c < in_f) ? npj[v,c] * deg[v] : 0
__global__ void tmp_build_kernel(const float* __restrict__ npj, const unsigned* __restrict__ deg,
                                 float* __restrict__ tmp, int N, int in_f, int tw)
{
  long long idx = (long long)blockIdx.x * blockDim.x + threadIdx.x;
  if (idx >= (long long)N * tw) return;
  int n = (int)(idx / tw), c = (int)(idx % tw);
  tmp[idx] = (c < in_f) ? npj[(size_t)n*in_f + c] * (float)deg[n] : 0.f;
}

// tmp[src[e], in_f+j] += e_mlp[e,j];  tmp[dst[e], in_f+j] += e_mlp[e,j]
__global__ void edge_scatter_kernel(const float* __restrict__ emlp,
                                    const int* __restrict__ src, const int* __restrict__ dst,
                                    float* __restrict__ tmp, int E, int in_f, int tw)
{
  long long idx = (long long)blockIdx.x * blockDim.x + threadIdx.x;
  if (idx >= (long long)E * 3) return;
  int e = (int)(idx / 3), j = (int)(idx % 3);
  float v = emlp[idx];
  atomicAdd(&tmp[(size_t)src[e]*tw + in_f + j], v);
  atomicAdd(&tmp[(size_t)dst[e]*tw + in_f + j], v);
}

// el[n,h] = sum_o feat[n,h,o]*attn_l[h,o];  er likewise
__global__ void attn_lr_kernel(const float* __restrict__ feat,
                               const float* __restrict__ al, const float* __restrict__ ar,
                               float* __restrict__ el, float* __restrict__ er,
                               int N, int out_f)
{
  int idx = blockIdx.x * blockDim.x + threadIdx.x;
  if (idx >= N * NHEAD) return;
  int n = idx / NHEAD, h = idx % NHEAD;
  const int HOF = NHEAD * out_f;
  const float* fr = feat + (size_t)n*HOF + h*out_f;
  const float* lw = al + h*out_f;
  const float* rw = ar + h*out_f;
  float sl = 0.f, sr = 0.f;
  for (int o = 0; o < out_f; ++o) { float f = fr[o]; sl += f*lw[o]; sr += f*rw[o]; }
  el[idx] = sl; er[idx] = sr;
}

// lg = leaky_relu(el[src]+er[dst], 0.2); segment max over dst (encoded atomicMax)
__global__ void lg_kernel(const float* __restrict__ el, const float* __restrict__ er,
                          const int* __restrict__ src, const int* __restrict__ dst,
                          float* __restrict__ lg, unsigned* __restrict__ menc, int E)
{
  long long idx = (long long)blockIdx.x * blockDim.x + threadIdx.x;
  if (idx >= (long long)E * NHEAD) return;
  int e = (int)(idx / NHEAD), h = (int)(idx % NHEAD);
  float v = el[(size_t)src[e]*NHEAD + h] + er[(size_t)dst[e]*NHEAD + h];
  v = v > 0.f ? v : 0.2f * v;
  lg[idx] = v;
  atomicMax(&menc[(size_t)dst[e]*NHEAD + h], fenc(v));
}

// a = exp(lg - m[dst]);  s[dst] += a
__global__ void alpha_kernel(const float* __restrict__ lg, const unsigned* __restrict__ menc,
                             const int* __restrict__ dst,
                             float* __restrict__ alpha, float* __restrict__ s, int E)
{
  long long idx = (long long)blockIdx.x * blockDim.x + threadIdx.x;
  if (idx >= (long long)E * NHEAD) return;
  int e = (int)(idx / NHEAD), h = (int)(idx % NHEAD);
  float m = fdec(menc[(size_t)dst[e]*NHEAD + h]);
  float a = expf(lg[idx] - m);
  alpha[idx] = a;
  atomicAdd(&s[(size_t)dst[e]*NHEAD + h], a);
}

__global__ void alpha_norm_kernel(float* __restrict__ alpha, const float* __restrict__ s,
                                  const int* __restrict__ dst, int E)
{
  long long idx = (long long)blockIdx.x * blockDim.x + threadIdx.x;
  if (idx >= (long long)E * NHEAD) return;
  int e = (int)(idx / NHEAD), h = (int)(idx % NHEAD);
  alpha[idx] /= s[(size_t)dst[e]*NHEAD + h];
}

// agg[dst, f] += alpha[e, f/out_f] * feat[src, f]
__global__ void aggregate_kernel(const float* __restrict__ alpha, const float* __restrict__ feat,
                                 const int* __restrict__ src, const int* __restrict__ dst,
                                 float* __restrict__ agg, int E, int HOF, int out_f)
{
  long long idx = (long long)blockIdx.x * blockDim.x + threadIdx.x;
  if (idx >= (long long)E * HOF) return;
  int e = (int)(idx / HOF), f = (int)(idx % HOF);
  int h = f / out_f;
  float a = alpha[(size_t)e*NHEAD + h];
  float v = feat[(size_t)src[e]*HOF + f];
  atomicAdd(&agg[(size_t)dst[e]*HOF + f], a * v);
}

// h_out = elu(agg + bias)
__global__ void bias_elu_kernel(const float* __restrict__ agg, const float* __restrict__ bias,
                                float* __restrict__ out, int N, int HOF)
{
  long long idx = (long long)blockIdx.x * blockDim.x + threadIdx.x;
  if (idx >= (long long)N * HOF) return;
  int f = (int)(idx % HOF);
  float v = agg[idx] + bias[f];
  out[idx] = v > 0.f ? v : (expf(v) - 1.f);
}

// gate[n] = poolhid[n,:] . pool_w2 + pool_b2
__global__ void gate_kernel(const float* __restrict__ ph, const float* __restrict__ w2,
                            const float* __restrict__ b2, float* __restrict__ gate, int N)
{
  int n = blockIdx.x * blockDim.x + threadIdx.x;
  if (n >= N) return;
  float a = b2[0];
  const float* r = ph + (size_t)n*128;
  for (int k = 0; k < 128; ++k) a += r[k]*w2[k];
  gate[n] = a;
}

__global__ void reduce_max_kernel(const float* __restrict__ g, unsigned* __restrict__ enc, int N)
{
  __shared__ float sm[256];
  int tid = threadIdx.x;
  int i = blockIdx.x * 256 + tid;
  sm[tid] = (i < N) ? g[i] : -3.0e38f;
  __syncthreads();
  for (int s = 128; s > 0; s >>= 1) { if (tid < s) sm[tid] = fmaxf(sm[tid], sm[tid+s]); __syncthreads(); }
  if (tid == 0) atomicMax(enc, fenc(sm[0]));
}

__global__ void softmax_w_kernel(const float* __restrict__ gate, const unsigned* __restrict__ enc,
                                 float* __restrict__ w, float* __restrict__ wsum, int N)
{
  int n = blockIdx.x * blockDim.x + threadIdx.x;
  if (n >= N) return;
  float v = expf(gate[n] - fdec(*enc));
  w[n] = v;
  atomicAdd(wsum, v);
}

// ge[f] = sum_n w[n]*emb[n,f] / wsum      (one block per feature)
__global__ void graph_emb_kernel(const float* __restrict__ w, const float* __restrict__ wsum,
                                 const float* __restrict__ emb, float* __restrict__ ge, int N)
{
  int f = blockIdx.x, tid = threadIdx.x;
  float acc = 0.f;
  for (int n = tid; n < N; n += 256) acc += w[n]*emb[(size_t)n*1024 + f];
  __shared__ float sm[256];
  sm[tid] = acc; __syncthreads();
  for (int s = 128; s > 0; s >>= 1) { if (tid < s) sm[tid] += sm[tid+s]; __syncthreads(); }
  if (tid == 0) ge[f] = sm[0] / wsum[0];
}

// ---------------------------------------------------------------------------
// VAE head: all the [1, *] math in one block.
// ---------------------------------------------------------------------------
struct HeadP {
  const float *ge, *lattice, *eps;
  const float *lat_w, *lat_b, *mu_w, *mu_b, *lv_w, *lv_b, *lp_w, *lp_b;
  const float *ed_w1, *ed_b1, *ed_w2, *ed_b2;
  const float *en_w1, *en_b1, *en_w2, *en_b2;
  const float *st_w1, *st_b1, *st_w2, *st_b2;
  float *out_mu, *out_lv, *out_z, *out_en, *out_st;
  float *zproj, *eout;
};

__global__ void __launch_bounds__(256) small_head_kernel(HeadP p)
{
  __shared__ float gsh[1152];   // g_emb = [graph_emb(1024), lat_proj(128)]
  __shared__ float zsh[64];
  __shared__ float zp[256];
  __shared__ float hid[128];
  const int tid = threadIdx.x;

  for (int i = tid; i < 1024; i += 256) gsh[i] = p.ge[i];
  for (int j = tid; j < 128; j += 256) {
    float a = p.lat_b[j];
    for (int k = 0; k < 6; ++k) a += p.lattice[k]*p.lat_w[k*128 + j];
    gsh[1024 + j] = a;
  }
  __syncthreads();

  // mu / logvar / z
  for (int j = tid; j < 64; j += 256) {
    float mu = p.mu_b[j], lv = p.lv_b[j];
    for (int k = 0; k < 1152; ++k) { float g = gsh[k]; mu += g*p.mu_w[k*64 + j]; lv += g*p.lv_w[k*64 + j]; }
    lv = fminf(10.f, fmaxf(-10.f, lv));
    float sd = fminf(1e6f, fmaxf(1e-6f, expf(0.5f*lv)));
    float z = mu + p.eps[j]*sd;
    p.out_mu[j] = mu; p.out_lv[j] = lv; p.out_z[j] = z; zsh[j] = z;
  }
  __syncthreads();

  // z_proj = relu(z @ lp_w + lp_b)
  for (int j = tid; j < 256; j += 256) {
    float a = p.lp_b[j];
    for (int k = 0; k < 64; ++k) a += zsh[k]*p.lp_w[k*256 + j];
    a = a > 0.f ? a : 0.f;
    zp[j] = a; p.zproj[j] = a;
  }
  __syncthreads();

  // recon_edge row (rank-1: z_exp[src]==z_exp[dst]==z_proj for every edge)
  for (int j = tid; j < 128; j += 256) {
    float a = p.ed_b1[j];
    for (int k = 0; k < 256; ++k) a += zp[k]*(p.ed_w1[k*128 + j] + p.ed_w1[(256 + k)*128 + j]);
    hid[j] = a > 0.f ? a : 0.f;
  }
  __syncthreads();
  for (int j = tid; j < 3; j += 256) {
    float a = p.ed_b2[j];
    for (int k = 0; k < 128; ++k) a += hid[k]*p.ed_w2[k*3 + j];
    p.eout[j] = a;
  }
  __syncthreads();

  // pred_energy: zl = [z(64), lattice(6)]
  for (int j = tid; j < 128; j += 256) {
    float a = p.en_b1[j];
    for (int k = 0; k < 64; ++k) a += zsh[k]*p.en_w1[k*128 + j];
    for (int k = 0; k < 6; ++k)  a += p.lattice[k]*p.en_w1[(64 + k)*128 + j];
    hid[j] = a > 0.f ? a : 0.f;
  }
  __syncthreads();
  for (int j = tid; j < 2; j += 256) {
    float a = p.en_b2[j];
    for (int k = 0; k < 128; ++k) a += hid[k]*p.en_w2[k*2 + j];
    p.out_en[j] = a;
  }
  __syncthreads();

  // pred_stress
  for (int j = tid; j < 128; j += 256) {
    float a = p.st_b1[j];
    for (int k = 0; k < 64; ++k) a += zsh[k]*p.st_w1[k*128 + j];
    for (int k = 0; k < 6; ++k)  a += p.lattice[k]*p.st_w1[(64 + k)*128 + j];
    hid[j] = a > 0.f ? a : 0.f;
  }
  __syncthreads();
  for (int j = tid; j < 9; j += 256) {
    float a = p.st_b2[j];
    for (int k = 0; k < 128; ++k) a += hid[k]*p.st_w2[k*9 + j];
    p.out_st[j] = a;
  }
}

__global__ void broadcast_edge_kernel(const float* __restrict__ eout, float* __restrict__ out, int E)
{
  long long idx = (long long)blockIdx.x * blockDim.x + threadIdx.x;
  if (idx >= (long long)E * 3) return;
  out[idx] = eout[idx % 3];
}

// ---------------------------------------------------------------------------
// Host-side orchestration
// ---------------------------------------------------------------------------

// input index map (recursive insertion-order flattening of setup_inputs())
enum {
  I_NODE = 0, I_EDGE, I_SRC, I_DST, I_LATTICE, I_EPS,
  I_G1 = 6, I_G2 = 16, I_G3 = 26,   // each GAT: em_w1,em_b1,em_w2,em_b2,np_w,np_b,fc_w,attn_l,attn_r,bias
  I_POOL_W1 = 36, I_POOL_B1, I_POOL_W2, I_POOL_B2,
  I_LAT_W = 40, I_LAT_B,
  I_MU_W = 42, I_MU_B,
  I_LV_W = 44, I_LV_B,
  I_LP_W = 46, I_LP_B,
  I_NEP_W = 48, I_NEP_B,
  I_ND_W1 = 50, I_ND_B1, I_ND_W2, I_ND_B2,
  I_ED_W1 = 54, I_ED_B1, I_ED_W2, I_ED_B2,
  I_EN_W1 = 58, I_EN_B1, I_EN_W2, I_EN_B2,
  I_ST_W1 = 62, I_ST_B1, I_ST_W2, I_ST_B2
};

struct Ctx {
  hipStream_t st;
  int N, E;
  const float* ef;                 // edge features
  const int *src, *dst;
  float *agg, *feat, *tmp, *npj, *h, *emlp, *el, *er, *lg, *alpha, *s, *gate, *w, *ge, *zproj, *eout, *wsum;
  unsigned *menc, *deg, *gmaxe;
  unsigned short *abf, *bbf;       // bf16 GEMM operand staging
};

// C[M,N] = post + act(A[M,K] @ B[K,N] + bias)
static void gemm(const Ctx& c, const float* A, const float* B, const float* bias,
                 const float* post, float* C, int M, int N, int K,
                 int lda, int ldb, int ldc, int act)
{
  const int Mp = rup32(M), Np = rup32(N), Kp = rup32(K);
  cvt_a_kernel<<<gsz((long long)Mp*Kp,256),256,0,c.st>>>(A, c.abf, M, K, lda, Kp, (long long)Mp*Kp);
  cvt_b_kernel<<<gsz((long long)Np*Kp,256),256,0,c.st>>>(B, c.bbf, K, N, ldb, Kp, (long long)Np*Kp);
  dim3 g(Mp/32, Np/32);
  wmma_gemm_bf<<<g,32,0,c.st>>>(c.abf, c.bbf, bias, post, C, M, N, Kp, ldc, act);
}

static void run_gat_layer(const Ctx& c, const float* h_in, int in_f, int out_f,
                          const float* const* gw, float* h_out)
{
  const int N = c.N, E = c.E;
  const int HOF = NHEAD * out_f;
  const int tw  = in_f + EFDIM;
  hipStream_t st = c.st;

  edge_mlp_kernel<<<gsz(E,256),256,0,st>>>(c.ef, gw[0], gw[1], gw[2], gw[3], c.emlp, E);
  fill_u32<<<gsz(N,256),256,0,st>>>(c.deg, 0u, N);
  degree_kernel<<<gsz(E,256),256,0,st>>>(c.src, c.dst, c.deg, E);

  // npj = h_in @ np_w + np_b
  gemm(c, h_in, gw[4], gw[5], nullptr, c.npj, N, in_f, in_f, in_f, in_f, in_f, 0);

  tmp_build_kernel<<<gsz((long long)N*tw,256),256,0,st>>>(c.npj, c.deg, c.tmp, N, in_f, tw);
  edge_scatter_kernel<<<gsz((long long)E*3,256),256,0,st>>>(c.emlp, c.src, c.dst, c.tmp, E, in_f, tw);

  // feat = tmp @ fc_w
  gemm(c, c.tmp, gw[6], nullptr, nullptr, c.feat, N, HOF, tw, tw, HOF, HOF, 0);

  attn_lr_kernel<<<gsz((long long)N*NHEAD,256),256,0,st>>>(c.feat, gw[7], gw[8], c.el, c.er, N, out_f);

  fill_u32<<<gsz((long long)N*NHEAD,256),256,0,st>>>(c.menc, fenc(-3.0e38f), (long long)N*NHEAD);
  fill_f32<<<gsz((long long)N*NHEAD,256),256,0,st>>>(c.s, 0.f, (long long)N*NHEAD);
  lg_kernel<<<gsz((long long)E*NHEAD,256),256,0,st>>>(c.el, c.er, c.src, c.dst, c.lg, c.menc, E);
  alpha_kernel<<<gsz((long long)E*NHEAD,256),256,0,st>>>(c.lg, c.menc, c.dst, c.alpha, c.s, E);
  alpha_norm_kernel<<<gsz((long long)E*NHEAD,256),256,0,st>>>(c.alpha, c.s, c.dst, E);

  fill_f32<<<gsz((long long)N*HOF,256),256,0,st>>>(c.agg, 0.f, (long long)N*HOF);
  aggregate_kernel<<<gsz((long long)E*HOF,256),256,0,st>>>(c.alpha, c.feat, c.src, c.dst, c.agg, E, HOF, out_f);
  bias_elu_kernel<<<gsz((long long)N*HOF,256),256,0,st>>>(c.agg, gw[9], h_out, N, HOF);
}

extern "C" void kernel_launch(void* const* d_in, const int* in_sizes, int n_in,
                              void* d_out, int out_size, void* d_ws, size_t ws_size,
                              hipStream_t stream)
{
  (void)n_in; (void)out_size; (void)ws_size;
  const int N = in_sizes[I_NODE] / NFD;      // 10000
  const int E = in_sizes[I_SRC];             // 120000

  const float* F[66];
  for (int i = 0; i < 66; ++i) F[i] = (const float*)d_in[i];

  Ctx c;
  c.st  = stream; c.N = N; c.E = E;
  c.ef  = F[I_EDGE];
  c.src = (const int*)d_in[I_SRC];
  c.dst = (const int*)d_in[I_DST];

  // workspace layout (float units; bf16 staging buffers counted as half floats)
  float* ws = (float*)d_ws;
  size_t off = 0;
  c.agg   = ws + off; off += (size_t)N * 1024;       // aggregation accumulator
  c.feat  = ws + off; off += (size_t)N * 1024;       // projected features (also ndhid later)
  c.tmp   = ws + off; off += (size_t)N * 520;        // fused node+edge buffer (also x later)
  c.npj   = ws + off; off += (size_t)N * 512;        // node projection (also poolhid later)
  c.h     = ws + off; off += (size_t)N * 512;        // inter-layer hidden h1/h2
  c.abf   = (unsigned short*)(ws + off); off += ((size_t)rup32(N) * 1024) / 2;  // A bf16 [Mp<=10016, Kp<=1024]
  c.bbf   = (unsigned short*)(ws + off); off += ((size_t)1024 * 1024) / 2;      // B bf16 [Np<=1024, Kp<=1024]
  c.emlp  = ws + off; off += (size_t)E * 3;
  c.el    = ws + off; off += (size_t)N * NHEAD;
  c.er    = ws + off; off += (size_t)N * NHEAD;
  c.lg    = ws + off; off += (size_t)E * NHEAD;
  c.alpha = ws + off; off += (size_t)E * NHEAD;
  c.menc  = (unsigned*)(ws + off); off += (size_t)N * NHEAD;
  c.s     = ws + off; off += (size_t)N * NHEAD;
  c.deg   = (unsigned*)(ws + off); off += (size_t)N;
  c.gate  = ws + off; off += (size_t)N;
  c.w     = ws + off; off += (size_t)N;
  c.ge    = ws + off; off += 1024;
  c.gmaxe = (unsigned*)(ws + off); off += 1;
  c.wsum  = ws + off; off += 1;
  c.zproj = ws + off; off += 256;
  c.eout  = ws + off; off += 4;

  // output slices (flat, reference return order)
  float* out      = (float*)d_out;
  float* o_mu     = out;
  float* o_lv     = out + 64;
  float* o_z      = out + 128;
  float* o_emb    = out + 192;
  float* o_rnode  = o_emb   + (size_t)N * 1024;
  float* o_redge  = o_rnode + (size_t)N * 4;
  float* o_en     = o_redge + (size_t)E * 3;
  float* o_st     = o_en + 2;

  // ---- 3 GAT layers (layer 3 ELU output written straight into d_out node_emb) ----
  run_gat_layer(c, F[I_NODE], NFD, 64,  &F[I_G1], c.h);     // h1: [N,256]
  run_gat_layer(c, c.h,       256, 128, &F[I_G2], c.h);     // h2: [N,512] (overwrites h1 after last read)
  run_gat_layer(c, c.h,       512, 256, &F[I_G3], o_emb);   // node_emb: [N,1024]

  // ---- Global attention pooling ----
  // poolhid = relu(emb @ pool_w1 + pool_b1)  -> reuse npj buffer [N,128]
  gemm(c, o_emb, F[I_POOL_W1], F[I_POOL_B1], nullptr, c.npj, N, 128, 1024, 1024, 128, 128, 1);
  gate_kernel<<<gsz(N,256),256,0,stream>>>(c.npj, F[I_POOL_W2], F[I_POOL_B2], c.gate, N);
  fill_u32<<<1,32,0,stream>>>(c.gmaxe, fenc(-3.0e38f), 1);
  reduce_max_kernel<<<gsz(N,256),256,0,stream>>>(c.gate, c.gmaxe, N);
  fill_f32<<<1,32,0,stream>>>(c.wsum, 0.f, 1);
  softmax_w_kernel<<<gsz(N,256),256,0,stream>>>(c.gate, c.gmaxe, c.w, c.wsum, N);
  graph_emb_kernel<<<1024,256,0,stream>>>(c.w, c.wsum, o_emb, c.ge, N);

  // ---- VAE head (mu/logvar/z, z_proj, rank-1 recon_edge row, energy, stress) ----
  HeadP hp;
  hp.ge = c.ge; hp.lattice = F[I_LATTICE]; hp.eps = F[I_EPS];
  hp.lat_w = F[I_LAT_W]; hp.lat_b = F[I_LAT_B];
  hp.mu_w = F[I_MU_W]; hp.mu_b = F[I_MU_B];
  hp.lv_w = F[I_LV_W]; hp.lv_b = F[I_LV_B];
  hp.lp_w = F[I_LP_W]; hp.lp_b = F[I_LP_B];
  hp.ed_w1 = F[I_ED_W1]; hp.ed_b1 = F[I_ED_B1]; hp.ed_w2 = F[I_ED_W2]; hp.ed_b2 = F[I_ED_B2];
  hp.en_w1 = F[I_EN_W1]; hp.en_b1 = F[I_EN_B1]; hp.en_w2 = F[I_EN_W2]; hp.en_b2 = F[I_EN_B2];
  hp.st_w1 = F[I_ST_W1]; hp.st_b1 = F[I_ST_B1]; hp.st_w2 = F[I_ST_W2]; hp.st_b2 = F[I_ST_B2];
  hp.out_mu = o_mu; hp.out_lv = o_lv; hp.out_z = o_z; hp.out_en = o_en; hp.out_st = o_st;
  hp.zproj = c.zproj; hp.eout = c.eout;
  small_head_kernel<<<1,256,0,stream>>>(hp);

  // ---- Node decoder ----
  // x = relu(emb @ nep_w + nep_b) + z_proj   -> reuse tmp buffer [N,256]
  gemm(c, o_emb, F[I_NEP_W], F[I_NEP_B], c.zproj, c.tmp, N, 256, 1024, 1024, 256, 256, 1);
  // ndhid = relu(x @ nd_w1 + nd_b1)          -> reuse feat buffer [N,128]
  gemm(c, c.tmp, F[I_ND_W1], F[I_ND_B1], nullptr, c.feat, N, 128, 256, 256, 128, 128, 1);
  // recon_node = ndhid @ nd_w2 + nd_b2       -> d_out slice [N,4]
  gemm(c, c.feat, F[I_ND_W2], F[I_ND_B2], nullptr, o_rnode, N, 4, 128, 128, 4, 4, 0);

  // ---- Edge decoder: broadcast the single rank-1 row ----
  broadcast_edge_kernel<<<gsz((long long)E*3,256),256,0,stream>>>(c.eout, o_redge, E);
}
